// LambdaLayer_69329362092493
// MI455X (gfx1250) — matrix operand order repaired and checked by reading, not compiled, not run on hardware
//
#include <hip/hip_runtime.h>

// ---------------------------------------------------------------------------
// LambdaLayer for MI455X (gfx1250, wave32, WMMA).
// Heavy op: 23x23 conv (35.5 GFLOP). Decomposed into 23 row-convs: for each
// dy a K=32 GEMM (dx 0..22 zero-padded to 32) with A = emb[dy] (16x32 bf16)
// and B = 16 CONSECUTIVE bf16 per lane from a zero-padded LDS value map.
// 8 flat-shifted LDS copies of the map guarantee 16B alignment, so every
// B fragment is exactly 2 ds_load_b128 (immediate offsets, no per-elem VALU).
// Precision-sensitive small ops (BN/softmax/lambda_c/y_c/y_p) stay f32 VALU.
// ---------------------------------------------------------------------------

typedef __attribute__((ext_vector_type(16))) __bf16        v16bf;
typedef __attribute__((ext_vector_type(8)))  float         v8f;
typedef __attribute__((ext_vector_type(4)))  unsigned int  u32x4;

#define BB   32
#define CC   256
#define HH   32
#define WW   32
#define NN   1024      // H*W
#define KD   16        // depth k
#define NH   4         // heads
#define VV   64        // dim_v
#define RK   23        // kernel R
#define PD   11        // pad
#define R2   529       // R*R
#define OCH  144       // 64 q + 16 k + 64 v fused projection
#define EPSS 1e-5f

#define PRW  64        // padded row stride (cols 0..63; data at 11..42)
#define PRH  54        // padded rows (rows 0..53; data at 11..42)
#define CPS  (PRH * PRW)   // 3456 elems per map copy
#define NCP  8             // shifted copies (16B alignment for any start)
#define EB2  (RK * KD * 32)  // 11776 elems: emb re-layout [dy][k][dx<32]

union Frag16 {
    v16bf          v;
    u32x4          q[2];
    __bf16         h[16];
    unsigned short u[16];
};

// ---------------- ws layout (bytes, all 256-aligned) ----------------
#define OFF_XBF   0u            // [b][n][c] bf16 : 16,777,216
#define OFF_WBF   16777216u     // [144][256] bf16: 73,728
#define OFF_EBF   16850944u     // [23][16][32] bf16 : 23,552
#define OFF_QRAW  16874496u     // [b][64][n] f32 : 8,388,608  (BN'd in place -> queries)
#define OFF_KRAW  25263104u     // [b][16][n] f32 : 2,097,152  (softmax'd in place)
#define OFF_VRAW  27360256u     // [b][64][n] f32 : 8,388,608  (BN'd in place -> values)
#define OFF_VBF   35748864u     // [b][64][n] bf16: 4,194,304
#define OFF_STATS 39943168u     // 256 f32 (qmean,qistd,vmean,vistd)
#define OFF_LC    39944192u     // [b][16][64] f32: 131,072
// total ~40.1 MB

// ---------------- K0a: x[b][c][n] f32 -> xbf[b][n][c] bf16 ----------------
__global__ void k_cvt_x(const float* __restrict__ x, __bf16* __restrict__ xbf) {
    const int total = BB * NN * CC;
    for (int i = blockIdx.x * blockDim.x + threadIdx.x; i < total;
         i += gridDim.x * blockDim.x) {
        int c = i & (CC - 1);
        int n = (i >> 8) & (NN - 1);
        int b = i >> 18;
        xbf[i] = (__bf16)x[(b * CC + c) * NN + n];
    }
}

// ------- K0b: fused weights bf16 + embedding re-layout [dy][k][dx<32] -----
__global__ void k_cvt_we(const float* __restrict__ Wq, const float* __restrict__ Wk,
                         const float* __restrict__ Wv, const float* __restrict__ emb,
                         __bf16* __restrict__ wbf, __bf16* __restrict__ ebf2) {
    const int nw = OCH * CC;            // 36864
    const int total = nw + EB2;         // + 11776
    for (int i = blockIdx.x * blockDim.x + threadIdx.x; i < total;
         i += gridDim.x * blockDim.x) {
        if (i < nw) {
            int c = i & (CC - 1);
            int oc = i >> 8;
            float v;
            if (oc < 64)       v = Wq[oc * CC + c];
            else if (oc < 80)  v = Wk[(oc - 64) * CC + c];
            else               v = Wv[(oc - 80) * CC + c];
            wbf[i] = (__bf16)v;
        } else {
            int j  = i - nw;           // [dy][k][dx]
            int dx = j & 31;
            int k  = (j >> 5) & 15;
            int dy = j >> 9;
            ebf2[j] = (dx < RK) ? (__bf16)emb[k * R2 + dy * RK + dx] : (__bf16)0.0f;
        }
    }
}

// ---------------- K1: fused qkv projection, WMMA bf16 GEMM ----------------
// per batch: [144 x 256] x [256 x 1024], M-tile per block, 4 waves split N.
__global__ __launch_bounds__(128) void k_proj(const __bf16* __restrict__ xbf,
                                              const __bf16* __restrict__ wbf,
                                              float* __restrict__ qraw,
                                              float* __restrict__ kraw,
                                              float* __restrict__ vraw) {
    __shared__ __attribute__((aligned(16))) __bf16 wt[16][CC];  // 8 KB
    const int b = blockIdx.x, mt = blockIdx.y;
    const int tid = threadIdx.x;
    for (int i = tid; i < 16 * CC; i += 128) {
        int row = i >> 8, c = i & (CC - 1);
        wt[row][c] = wbf[(mt * 16 + row) * CC + c];
    }
    __syncthreads();

    const int wave = tid >> 5, lane = tid & 31;
    const int m     = lane & 15;
    const int kloA  = (lane < 16) ? 0 : 16;   // A: 16 contiguous K
    const int kloB  = (lane < 16) ? 0 : 16;   // B: 16 contiguous K

    for (int nt = wave; nt < 64; nt += 4) {
        const int n = nt * 16 + m;
        const __bf16* xrow = xbf + ((size_t)(b * NN + n)) * CC;
        v8f acc = {};
#pragma unroll
        for (int kc = 0; kc < 8; ++kc) {
            Frag16 a, fb;
            const int ka = kc * 32 + kloA;
            a.q[0] = *(const u32x4*)&wt[m][ka];
            a.q[1] = *(const u32x4*)&wt[m][ka + 8];
            const int kb = kc * 32 + kloB;
            fb.q[0] = *(const u32x4*)&xrow[kb];
            fb.q[1] = *(const u32x4*)&xrow[kb + 8];
            acc = __builtin_amdgcn_wmma_f32_16x16x32_bf16(
                false, a.v, false, fb.v, (short)0, acc, false, false);
        }
        const int nidx = nt * 16 + (lane & 15);
        const int mh   = (lane >> 4) * 8;
#pragma unroll
        for (int j = 0; j < 8; ++j) {
            const int oc = mt * 16 + j + mh;
            const float val = acc[j];
            if (oc < 64)       qraw[(b * 64 + oc) * NN + nidx] = val;
            else if (oc < 80)  kraw[(b * 16 + (oc - 64)) * NN + nidx] = val;
            else               vraw[(b * 64 + (oc - 80)) * NN + nidx] = val;
        }
    }
}

// ---------------- K2: BN batch stats (block per channel) ------------------
__global__ __launch_bounds__(256) void k_stats(const float* __restrict__ qraw,
                                               const float* __restrict__ vraw,
                                               float* __restrict__ stats) {
    __shared__ float s1[256], s2[256];
    const int ch = blockIdx.x;                 // 0..63 q, 64..127 v
    const float* base = (ch < 64) ? qraw : vraw;
    const int c = ch & 63;
    float sum = 0.f, sq = 0.f;
    for (int i = threadIdx.x; i < BB * NN; i += 256) {
        int b = i >> 10, n = i & (NN - 1);
        float v = base[(b * 64 + c) * NN + n];
        sum += v;
        sq  += v * v;
    }
    s1[threadIdx.x] = sum;
    s2[threadIdx.x] = sq;
    __syncthreads();
    for (int s = 128; s > 0; s >>= 1) {
        if (threadIdx.x < s) {
            s1[threadIdx.x] += s1[threadIdx.x + s];
            s2[threadIdx.x] += s2[threadIdx.x + s];
        }
        __syncthreads();
    }
    if (threadIdx.x == 0) {
        const float inv = 1.0f / (float)(BB * NN);
        float mean = s1[0] * inv;
        float var  = s2[0] * inv - mean * mean;
        int o = (ch < 64) ? 0 : 128;
        stats[o + c]      = mean;
        stats[o + 64 + c] = rsqrtf(var + EPSS);
    }
}

// ---------------- K3a: apply BN to q,v (in place); bf16 copy of v ---------
__global__ void k_bn(float* __restrict__ qraw, float* __restrict__ vraw,
                     __bf16* __restrict__ vbf, const float* __restrict__ stats,
                     const float* __restrict__ bnqw, const float* __restrict__ bnqb,
                     const float* __restrict__ bnvw, const float* __restrict__ bnvb) {
    const int NQ = BB * 64 * NN;   // 2M each
    for (int i = blockIdx.x * blockDim.x + threadIdx.x; i < 2 * NQ;
         i += gridDim.x * blockDim.x) {
        if (i < NQ) {
            int ch = (i >> 10) & 63;
            float v = (qraw[i] - stats[ch]) * stats[64 + ch] * bnqw[ch] + bnqb[ch];
            qraw[i] = v;
        } else {
            int j = i - NQ;
            int ch = (j >> 10) & 63;
            float v = (vraw[j] - stats[128 + ch]) * stats[192 + ch] * bnvw[ch] + bnvb[ch];
            vraw[j] = v;
            vbf[j]  = (__bf16)v;
        }
    }
}

// ---------------- K3b: softmax over n, per (b,k), in place ----------------
__global__ __launch_bounds__(256) void k_softmax(float* __restrict__ kraw) {
    __shared__ float sr[256];
    float* p = kraw + (size_t)blockIdx.x * NN;
    float mx = -1e30f;
    for (int i = threadIdx.x; i < NN; i += 256) mx = fmaxf(mx, p[i]);
    sr[threadIdx.x] = mx;
    __syncthreads();
    for (int s = 128; s > 0; s >>= 1) {
        if (threadIdx.x < s) sr[threadIdx.x] = fmaxf(sr[threadIdx.x], sr[threadIdx.x + s]);
        __syncthreads();
    }
    mx = sr[0];
    __syncthreads();
    float sum = 0.f;
    for (int i = threadIdx.x; i < NN; i += 256) {
        float e = __expf(p[i] - mx);
        p[i] = e;
        sum += e;
    }
    sr[threadIdx.x] = sum;
    __syncthreads();
    for (int s = 128; s > 0; s >>= 1) {
        if (threadIdx.x < s) sr[threadIdx.x] += sr[threadIdx.x + s];
        __syncthreads();
    }
    const float inv = 1.0f / sr[0];
    for (int i = threadIdx.x; i < NN; i += 256) p[i] *= inv;
}

// ---------------- K4a: lambda_c[b,k,v] = sum_n sm * val (f32, LDS tiled) --
__global__ __launch_bounds__(1024) void k_lambdac(const float* __restrict__ kraw,
                                                  const float* __restrict__ vraw,
                                                  float* __restrict__ lc) {
    __shared__ float ssm[16][128];   // 8 KB
    __shared__ float sv[64][128];    // 32 KB
    const int b = blockIdx.x;
    const int t = threadIdx.x;
    const int k = t >> 6, v = t & 63;
    float acc = 0.f;
    for (int c0 = 0; c0 < NN; c0 += 128) {
        for (int i = t; i < 16 * 128; i += 1024)
            ssm[i >> 7][i & 127] = kraw[(b * 16 + (i >> 7)) * NN + c0 + (i & 127)];
        for (int i = t; i < 64 * 128; i += 1024)
            sv[i >> 7][i & 127] = vraw[(b * 64 + (i >> 7)) * NN + c0 + (i & 127)];
        __syncthreads();
#pragma unroll 8
        for (int j = 0; j < 128; ++j) acc += ssm[k][j] * sv[v][j];
        __syncthreads();
    }
    lc[(b * 16 + k) * 64 + v] = acc;
}

// ---------------- K4b: out = (1+gamma) * y_c  (writes every element) ------
__global__ __launch_bounds__(256) void k_yc(const float* __restrict__ qraw,
                                            const float* __restrict__ lc,
                                            const float* __restrict__ gamma,
                                            float* __restrict__ out) {
    __shared__ float slc[16 * 64];   // 4 KB
    const int b = blockIdx.x;
    const int n = blockIdx.y * 256 + threadIdx.x;
    for (int i = threadIdx.x; i < 1024; i += 256) slc[i] = lc[b * 1024 + i];
    __syncthreads();
    const float scale = 1.0f + gamma[0];
    for (int h = 0; h < NH; ++h) {
        float q[16];
#pragma unroll
        for (int k = 0; k < 16; ++k) q[k] = qraw[(b * 64 + h * 16 + k) * NN + n];
        for (int v = 0; v < VV; ++v) {
            float s = 0.f;
#pragma unroll
            for (int k = 0; k < 16; ++k) s += q[k] * slc[k * 64 + v];
            out[((size_t)(b * 256) + h * 64 + v) * NN + n] = scale * s;
        }
    }
}

// ---------------- K5: 23 row-conv WMMA GEMMs + fused y_p ------------------
// block per (b,v). For each dy (23), one K=32 wmma per N-tile: A = emb[dy]
// row-major 16x32; B lane (pixel px) = 16 consecutive bf16 of padded row
// py+dy starting at col px+kloB. 8 shifted map copies give 16B alignment:
// each B fragment = 2 ds_load_b128 (immediate offsets), zero per-elem VALU.
__global__ __launch_bounds__(256) void k_conv_yp(const __bf16* __restrict__ vbf,
                                                 const __bf16* __restrict__ ebf2,
                                                 const float* __restrict__ qraw,
                                                 const float* __restrict__ gamma,
                                                 float* __restrict__ out) {
    __shared__ __attribute__((aligned(16))) __bf16 semb[EB2];        // 23552 B
    __shared__ __attribute__((aligned(16))) __bf16 spadC[NCP * CPS]; // 55296 B
    __shared__ float stile[8][256];                                  // 8 KB
    const int bv = blockIdx.x;
    const int b = bv >> 6, v = bv & 63;
    const int tid = threadIdx.x;

    for (int i = tid; i < EB2; i += 256) semb[i] = ebf2[i];

    // copy 0: zero-padded value map (rows/cols 11..42 hold the 32x32 data)
    const __bf16* vrow = vbf + (size_t)bv * NN;
    for (int j = tid; j < CPS; j += 256) {
        int pr = j >> 6, pc = j & 63;
        int y = pr - PD, x = pc - PD;
        __bf16 val = (__bf16)0.0f;
        if ((unsigned)y < 32u && (unsigned)x < 32u) val = vrow[y * 32 + x];
        spadC[j] = val;
    }
    __syncthreads();
    // copies 1..7: flat shift by c (reads only copy 0 region -> no race)
    for (int j = tid; j < (NCP - 1) * CPS; j += 256) {
        int c  = 1 + j / CPS;
        int jj = j - (c - 1) * CPS;
        int src = jj + c;
        spadC[c * CPS + jj] = (src < CPS) ? spadC[src] : (__bf16)0.0f;
    }
    __syncthreads();

    const int wave = tid >> 5, lane = tid & 31;
    const int m    = lane & 15;
    const int kloA = (lane < 16) ? 0 : 16;
    const int kloB = (lane < 16) ? 0 : 16;
    const float scale = 1.0f + gamma[0];

    // B-fragment pixel for tile t: n0 = (wave + 8*t)*16, p = n0 + m.
    // 128*t == 0 mod 32: px t-invariant, py advances 4 rows (256 elems)/tile.
    const int p0  = wave * 16 + m;
    const int px  = p0 & 31;
    const int py0 = p0 >> 5;

    const int colc = px + kloB;            // start col in padded row
    const int csel = colc & 7;             // which shifted copy
    const __bf16* myc = &spadC[csel * CPS + (colc - csel)]; // 16B-aligned base

    v8f acc[8];
    {
        v8f z = {};
#pragma unroll
        for (int t = 0; t < 8; ++t) acc[t] = z;
    }

    for (int dy = 0; dy < RK; ++dy) {
        Frag16 a;
        const __bf16* arow = &semb[(dy * KD + m) * 32];
        a.q[0] = *(const u32x4*)&arow[kloA];
        a.q[1] = *(const u32x4*)&arow[kloA + 8];
        const int rowoff = (py0 + dy) * PRW;
#pragma unroll
        for (int t = 0; t < 8; ++t) {
            Frag16 fb;
            const __bf16* bp = myc + rowoff + t * 256;
            fb.q[0] = *(const u32x4*)bp;
            fb.q[1] = *(const u32x4*)(bp + 8);
            acc[t] = __builtin_amdgcn_wmma_f32_16x16x32_bf16(
                false, a.v, false, fb.v, (short)0, acc[t], false, false);
        }
    }

    // epilogue: per tile, stash lambda_p 16x16 and contract with queries.
    // (C layout: VGPR j -> M = j + 8*(lane>=16), N = lane&15)
    const int mh = (lane >> 4) * 8;
    for (int t = 0; t < 8; ++t) {
        const int n0 = (wave + 8 * t) * 16;
#pragma unroll
        for (int j = 0; j < 8; ++j)
            stile[wave][(j + mh) * 16 + (lane & 15)] = acc[t][j];
        // same-wave LDS store->load: DS ops are in-order per wave.
#pragma unroll
        for (int i = 0; i < 2; ++i) {
            const int o = i * 32 + lane;
            const int h = o >> 4, pp = o & 15;
            const int n = n0 + pp;
            float s = 0.f;
#pragma unroll
            for (int k = 0; k < 16; ++k)
                s += qraw[(b * 64 + h * 16 + k) * NN + n] * stile[wave][k * 16 + pp];
            const size_t idx = ((size_t)(b * 256) + h * 64 + v) * NN + n;
            out[idx] += scale * s;
        }
    }
}

// ---------------------------------------------------------------------------
extern "C" void kernel_launch(void* const* d_in, const int* in_sizes, int n_in,
                              void* d_out, int out_size, void* d_ws, size_t ws_size,
                              hipStream_t stream) {
    (void)in_sizes; (void)n_in; (void)out_size; (void)ws_size;
    const float* x     = (const float*)d_in[0];
    const float* Wq    = (const float*)d_in[1];
    const float* bnqw  = (const float*)d_in[2];
    const float* bnqb  = (const float*)d_in[3];
    const float* Wk    = (const float*)d_in[4];
    const float* Wv    = (const float*)d_in[5];
    const float* bnvw  = (const float*)d_in[6];
    const float* bnvb  = (const float*)d_in[7];
    const float* emb   = (const float*)d_in[8];
    const float* gamma = (const float*)d_in[9];
    float* out = (float*)d_out;

    char* ws = (char*)d_ws;
    __bf16* xbf   = (__bf16*)(ws + OFF_XBF);
    __bf16* wbf   = (__bf16*)(ws + OFF_WBF);
    __bf16* ebf2  = (__bf16*)(ws + OFF_EBF);
    float*  qraw  = (float*)(ws + OFF_QRAW);
    float*  kraw  = (float*)(ws + OFF_KRAW);
    float*  vraw  = (float*)(ws + OFF_VRAW);
    __bf16* vbf   = (__bf16*)(ws + OFF_VBF);
    float*  stats = (float*)(ws + OFF_STATS);
    float*  lc    = (float*)(ws + OFF_LC);

    k_cvt_x <<<2048, 256, 0, stream>>>(x, xbf);
    k_cvt_we<<<64,   256, 0, stream>>>(Wq, Wk, Wv, emb, wbf, ebf2);
    k_proj  <<<dim3(BB, OCH / 16), 128, 0, stream>>>(xbf, wbf, qraw, kraw, vraw);
    k_stats <<<128, 256, 0, stream>>>(qraw, vraw, stats);
    k_bn    <<<2048, 256, 0, stream>>>(qraw, vraw, vbf, stats, bnqw, bnqb, bnvw, bnvb);
    k_softmax<<<BB * KD, 256, 0, stream>>>(kraw);
    k_lambdac<<<BB, 1024, 0, stream>>>(kraw, vraw, lc);
    k_yc    <<<dim3(BB, 4), 256, 0, stream>>>(qraw, lc, gamma, out);
    k_conv_yp<<<BB * VV, 256, 0, stream>>>(vbf, ebf2, qraw, gamma, out);
}
